// GroupedQueryAttention_33517924778609
// MI455X (gfx1250) — matrix operand order repaired and checked by reading, not compiled, not run on hardware
//
#include <hip/hip_runtime.h>
#include <hip/hip_bf16.h>
#include <cmath>

typedef __attribute__((ext_vector_type(16))) __bf16 v16bf;
typedef __attribute__((ext_vector_type(8)))  float  v8f;

union FragU {
  unsigned int u[8];
  v16bf bf;
};

__device__ __forceinline__ unsigned short f2bf(float f) {
  unsigned int u = __float_as_uint(f);
  u += 0x7FFFu + ((u >> 16) & 1u);   // round-to-nearest-even
  return (unsigned short)(u >> 16);
}

// A-operand (16-bit, 16x32): lane holds row m=lane&15.
// lanes 0-15: V0..3 K=0..7, V4..7 K=16..23 ; lanes 16-31: K=8..15 / 24..31
__device__ __forceinline__ v16bf load_frag_a(const unsigned short* rowPtr, int laneHi) {
  FragU f;
#pragma unroll
  for (int j = 0; j < 8; ++j) {
    int k = laneHi * 8 + ((j & 4) << 2) + (j & 3) * 2;
    f.u[j] = *(const unsigned int*)(rowPtr + k);
  }
  return f.bf;
}

// B-operand (16-bit, 32x16): lane holds column n=lane&15.
// lanes 0-15: V0..3 K=0..7, V4..7 K=16..23 ; lanes 16-31: K=8..15 / 24..31
__device__ __forceinline__ v16bf load_frag_b(const unsigned short* colPtr, int laneHi) {
  FragU f;
#pragma unroll
  for (int j = 0; j < 8; ++j) {
    int k = laneHi * 8 + ((j & 4) << 2) + (j & 3) * 2;
    f.u[j] = *(const unsigned int*)(colPtr + k);
  }
  return f.bf;
}

// B-operand gather from a row-major [K][N] tile (transpose on the fly).
__device__ __forceinline__ v16bf load_frag_b_strided(const unsigned short* base, int col,
                                                     int stride, int laneHi) {
  FragU f;
#pragma unroll
  for (int j = 0; j < 8; ++j) {
    int k = laneHi * 8 + ((j & 4) << 2) + (j & 3) * 2;
    unsigned int lo = base[k * stride + col];
    unsigned int hi = base[(k + 1) * stride + col];
    f.u[j] = lo | (hi << 16);
  }
  return f.bf;
}

__device__ __forceinline__ v8f wmma_bf16(v16bf a, v16bf b, v8f c) {
  return __builtin_amdgcn_wmma_f32_16x16x32_bf16(false, a, false, b, (short)0, c,
                                                 false, false);
}

// Async copy 16 bytes global -> LDS (gfx1250 ASYNCcnt path).
__device__ __forceinline__ void async_copy_b128(unsigned ldsOff, const void* gaddr) {
  unsigned long long ga = (unsigned long long)(size_t)gaddr;
  asm volatile("global_load_async_to_lds_b128 %0, %1, off" :: "v"(ldsOff), "v"(ga) : "memory");
}
// ASYNCcnt completes in order: waiting <=2 after issuing 2 new copies
// guarantees the previous block's 2 copies have landed.
__device__ __forceinline__ void wait_async_le2() {
  asm volatile("s_wait_asynccnt 0x2" ::: "memory");
}
__device__ __forceinline__ void wait_async_le0() {
  asm volatile("s_wait_asynccnt 0x0" ::: "memory");
}

// ---------------------------------------------------------------------------
// Tiled bf16 WMMA GEMM: C[M,N] = A[M,K] * B[K,N]
// A: fp32 or bf16 (template), B: fp32 (converted on the fly), C: bf16 or fp32.
// Workgroup: 256 threads (8 waves), 128x128 tile, K-step 32.
// Wave (w) computes 32(M) x 64(N): wm = (w>>1)*32, wn = (w&1)*64.
// ---------------------------------------------------------------------------
template <bool A_BF16, bool OUT_BF16>
__global__ __launch_bounds__(256) void gemm_wmma_kernel(const void* __restrict__ Aptr,
                                                        const float* __restrict__ B,
                                                        void* __restrict__ Cptr,
                                                        int M, int N, int K) {
  __shared__ __align__(16) unsigned short As[128 * 32];   // [m][k]
  __shared__ __align__(16) unsigned short Bs[128 * 32];   // [n][k] (transposed)

  const int tid    = threadIdx.x;
  const int wave   = tid >> 5;
  const int lane   = tid & 31;
  const int laneHi = lane >> 4;
  const int lane15 = lane & 15;
  const int m0 = blockIdx.y * 128;
  const int n0 = blockIdx.x * 128;
  const int wm = (wave >> 1) * 32;
  const int wn = (wave & 1) * 64;

  v8f acc[2][4];
#pragma unroll
  for (int i = 0; i < 2; ++i)
#pragma unroll
    for (int j = 0; j < 4; ++j) acc[i][j] = (v8f){};

  for (int k0 = 0; k0 < K; k0 += 32) {
    // ---- Stage A tile 128x32 with vectorized loads ----
    if constexpr (!A_BF16) {
      const float* Af = (const float*)Aptr;
#pragma unroll
      for (int i = 0; i < 4; ++i) {
        int idx = tid + i * 256;           // 1024 float4 = 4096 floats
        int r   = idx >> 3;                // row 0..127
        int c4  = idx & 7;                 // float4 within the 32-wide chunk
        float4 v = ((const float4*)(Af + (size_t)(m0 + r) * K + k0))[c4];
        unsigned p0 = (unsigned)f2bf(v.x) | ((unsigned)f2bf(v.y) << 16);
        unsigned p1 = (unsigned)f2bf(v.z) | ((unsigned)f2bf(v.w) << 16);
        *(uint2*)&As[r * 32 + c4 * 4] = make_uint2(p0, p1);
      }
    } else {
      const unsigned short* Ab = (const unsigned short*)Aptr;
#pragma unroll
      for (int i = 0; i < 2; ++i) {
        int idx = tid + i * 256;           // 512 uint4 = 4096 bf16
        int r   = idx >> 2;
        int c8  = idx & 3;                 // 8 bf16 per uint4
        uint4 v = ((const uint4*)(Ab + (size_t)(m0 + r) * K + k0))[c8];
        *(uint4*)&As[r * 32 + c8 * 8] = v;
      }
    }
    // ---- Stage B tile 32x128 transposed; pack 2 K per ds_store_b32 ----
#pragma unroll
    for (int i = 0; i < 8; ++i) {
      int idx = tid + i * 256;             // 2048 pairs
      int kk  = (idx >> 7) * 2;            // 0,2,..,30
      int nn  = idx & 127;
      unsigned lo = f2bf(B[(size_t)(k0 + kk) * N + (n0 + nn)]);
      unsigned hi = f2bf(B[(size_t)(k0 + kk + 1) * N + (n0 + nn)]);
      *(unsigned*)&Bs[nn * 32 + kk] = lo | (hi << 16);
    }
    // Prefetch next K-step tiles into cache while we compute this one.
    if (k0 + 32 < K) {
      size_t esz = A_BF16 ? 2 : 4;
      __builtin_prefetch((const char*)Aptr +
                         ((size_t)(m0 + (tid >> 3)) * K + k0 + 32) * esz, 0, 3);
      __builtin_prefetch(&B[(size_t)(k0 + 32 + (tid >> 7)) * N + n0 + (tid & 127)], 0, 3);
    }
    __syncthreads();

    v16bf af[2], bf[4];
#pragma unroll
    for (int ms = 0; ms < 2; ++ms)
      af[ms] = load_frag_a(&As[(wm + 16 * ms + lane15) * 32], laneHi);
#pragma unroll
    for (int ns = 0; ns < 4; ++ns)
      bf[ns] = load_frag_b(&Bs[(wn + 16 * ns + lane15) * 32], laneHi);

#pragma unroll
    for (int ms = 0; ms < 2; ++ms)
#pragma unroll
      for (int ns = 0; ns < 4; ++ns)
        acc[ms][ns] = wmma_bf16(af[ms], bf[ns], acc[ms][ns]);

    __syncthreads();
  }

  // Store: C layout — VGPR r: row = r + 8*laneHi, col = lane15
#pragma unroll
  for (int ms = 0; ms < 2; ++ms)
#pragma unroll
    for (int ns = 0; ns < 4; ++ns)
#pragma unroll
      for (int r = 0; r < 8; ++r) {
        int row = m0 + wm + 16 * ms + laneHi * 8 + r;
        int col = n0 + wn + 16 * ns + lane15;
        if constexpr (OUT_BF16)
          ((unsigned short*)Cptr)[(size_t)row * N + col] = f2bf(acc[ms][ns][r]);
        else
          ((float*)Cptr)[(size_t)row * N + col] = acc[ms][ns][r];
      }
}

// ---------------------------------------------------------------------------
// Flash attention (causal, GQA group 4). One workgroup per (b, h, 128-row q tile).
// 8 waves x 16 q-rows. K/V 32-key blocks are software-pipelined through a
// ping-pong LDS buffer with global_load_async_to_lds_b128: block kb+1's DMA
// is issued before waiting (s_wait_asynccnt 0x2, in-order completion) so the
// next tile streams in under the WMMA + softmax of the current tile.
// Q,K,V,ctx are bf16 in workspace: Q/ctx [B,S,2048], K/V [B,S,512].
// ---------------------------------------------------------------------------
__global__ __launch_bounds__(256) void attn_wmma_kernel(const unsigned short* __restrict__ Q,
                                                        const unsigned short* __restrict__ Kb,
                                                        const unsigned short* __restrict__ Vb,
                                                        unsigned short* __restrict__ Ctx) {
  constexpr int S = 2048, E = 2048, KV = 512, D = 64;
  __shared__ __align__(16) unsigned short Ks[2][32 * 64];  // [buf][key][d]
  __shared__ __align__(16) unsigned short Vs[2][32 * 64];  // [buf][key][d]
  __shared__ __align__(16) unsigned short Pl[8][16 * 32];  // per-wave P scratch [row][key]

  const int tid    = threadIdx.x;
  const int wave   = tid >> 5;
  const int lane   = tid & 31;
  const int laneHi = lane >> 4;
  const int lane15 = lane & 15;

  const int bid = blockIdx.x;          // B*H*(S/128) = 2*32*16 = 1024
  const int qt  = bid & 15;
  const int h   = (bid >> 4) & 31;
  const int b   = bid >> 9;
  const int kvh = h >> 2;

  // Per-thread async staging slot: 16B each, covers each 4KB tile exactly.
  const int str = tid >> 3;            // key row 0..31
  const int stc = (tid & 7) * 8;       // element col 0,8,...,56
  unsigned ldsK[2], ldsV[2];
#pragma unroll
  for (int bfi = 0; bfi < 2; ++bfi) {
    ldsK[bfi] = (unsigned)(size_t)&Ks[bfi][str * 64 + stc];
    ldsV[bfi] = (unsigned)(size_t)&Vs[bfi][str * 64 + stc];
  }
  const size_t gstage0 = (size_t)(b * S + str) * KV + kvh * D + stc;

  const int nkb = qt * 4 + 4;  // key blocks of 32 up through q tile end

  // Prologue: kick off block 0 DMA, then overlap Q fragment loads with it.
  async_copy_b128(ldsK[0], Kb + gstage0);
  async_copy_b128(ldsV[0], Vb + gstage0);

  // Q fragments (A-operand): lane row = qt*128 + wave*16 + lane15, d in 2 chunks
  const int sq_row = qt * 128 + wave * 16 + lane15;
  const unsigned short* qbase = Q + (size_t)(b * S + sq_row) * E + h * D;
  v16bf qf0 = load_frag_a(qbase, laneHi);
  v16bf qf1 = load_frag_a(qbase + 32, laneHi);

  v8f o[4];
#pragma unroll
  for (int i = 0; i < 4; ++i) o[i] = (v8f){};
  float mrun[8], lrun[8];
#pragma unroll
  for (int r = 0; r < 8; ++r) { mrun[r] = -INFINITY; lrun[r] = 0.0f; }

  const float scale = 0.125f;  // 1/sqrt(64)

  for (int kb = 0; kb < nkb; ++kb) {
    const int cur = kb & 1;
    const bool hasNext = (kb + 1) < nkb;
    if (hasNext) {
      // Issue next block's DMA into the other buffer (safe: all waves passed
      // the end-of-iteration barrier since last reading it), then wait for
      // the current block's 2 oldest copies (in-order ASYNCcnt).
      const size_t gn = gstage0 + (size_t)(kb + 1) * 32 * KV;
      async_copy_b128(ldsK[cur ^ 1], Kb + gn);
      async_copy_b128(ldsV[cur ^ 1], Vb + gn);
      wait_async_le2();
    } else {
      wait_async_le0();
    }
    __syncthreads();   // (A) current K/V tiles visible to all waves

    const int sk0 = kb * 32;
    const unsigned short* KsC = &Ks[cur][0];
    const unsigned short* VsC = &Vs[cur][0];

    // S = Q K^T : two 16x16 key sub-tiles, contraction over d (2 x 32)
    v8f s[2];
#pragma unroll
    for (int t = 0; t < 2; ++t) {
      const unsigned short* kcol = &KsC[(16 * t + lane15) * 64];
      v16bf kf0 = load_frag_b(kcol, laneHi);
      v16bf kf1 = load_frag_b(kcol + 32, laneHi);
      v8f sa = (v8f){};
      sa = wmma_bf16(qf0, kf0, sa);
      sa = wmma_bf16(qf1, kf1, sa);
      s[t] = sa;
    }

    // Scale + causal mask
#pragma unroll
    for (int t = 0; t < 2; ++t)
#pragma unroll
      for (int r = 0; r < 8; ++r) {
        int srow = qt * 128 + wave * 16 + laneHi * 8 + r;
        int scol = sk0 + 16 * t + lane15;
        float v = s[t][r] * scale;
        s[t][r] = (scol > srow) ? -INFINITY : v;
      }

    // Online softmax: row elements live in one 16-lane half per VGPR r
    float corr[8];
#pragma unroll
    for (int r = 0; r < 8; ++r) {
      float rm = fmaxf(s[0][r], s[1][r]);
#pragma unroll
      for (int off = 8; off >= 1; off >>= 1)
        rm = fmaxf(rm, __shfl_xor(rm, off, 32));
      float nm = fmaxf(mrun[r], rm);
      float c  = __expf(mrun[r] - nm);
      mrun[r]  = nm;
      float rs = 0.0f;
#pragma unroll
      for (int t = 0; t < 2; ++t) {
        float p = __expf(s[t][r] - nm);
        s[t][r] = p;
        rs += p;
      }
#pragma unroll
      for (int off = 8; off >= 1; off >>= 1)
        rs += __shfl_xor(rs, off, 32);
      lrun[r] = lrun[r] * c + rs;
      corr[r] = c;
    }
#pragma unroll
    for (int ns = 0; ns < 4; ++ns)
#pragma unroll
      for (int r = 0; r < 8; ++r) o[ns][r] *= corr[r];

    // P: C-layout -> A-layout via per-wave LDS scratch
    unsigned short* pw = &Pl[wave][0];
#pragma unroll
    for (int t = 0; t < 2; ++t)
#pragma unroll
      for (int r = 0; r < 8; ++r)
        pw[(laneHi * 8 + r) * 32 + 16 * t + lane15] = f2bf(s[t][r]);
    __syncthreads();   // (B)

    v16bf pf = load_frag_a(&Pl[wave][lane15 * 32], laneHi);

    // O += P V : 4 d-tiles of 16, contraction over 32 keys.
    // V is [key][d] in LDS; gather transposed B-operand on the fly.
#pragma unroll
    for (int ns = 0; ns < 4; ++ns) {
      v16bf vf = load_frag_b_strided(VsC, 16 * ns + lane15, 64, laneHi);
      o[ns] = wmma_bf16(pf, vf, o[ns]);
    }
    __syncthreads();   // (C) all reads of this buffer done before it is reused
  }

  // Normalize and write context (bf16) [B,S,2048]
  float inv[8];
#pragma unroll
  for (int r = 0; r < 8; ++r) inv[r] = 1.0f / lrun[r];
#pragma unroll
  for (int ns = 0; ns < 4; ++ns)
#pragma unroll
    for (int r = 0; r < 8; ++r) {
      int row = qt * 128 + wave * 16 + laneHi * 8 + r;
      int col = h * D + 16 * ns + lane15;
      Ctx[(size_t)(b * S + row) * E + col] = f2bf(o[ns][r] * inv[r]);
    }
}

// ---------------------------------------------------------------------------
extern "C" void kernel_launch(void* const* d_in, const int* in_sizes, int n_in,
                              void* d_out, int out_size, void* d_ws, size_t ws_size,
                              hipStream_t stream) {
  (void)in_sizes; (void)n_in; (void)out_size; (void)ws_size;
  const float* x  = (const float*)d_in[0];
  const float* Wq = (const float*)d_in[1];
  const float* Wk = (const float*)d_in[2];
  const float* Wv = (const float*)d_in[3];
  const float* Wo = (const float*)d_in[4];
  float* out = (float*)d_out;

  const int M = 2 * 2048;   // B*S = 4096
  const int E = 2048, KV = 512;

  char* ws = (char*)d_ws;
  unsigned short* Qp  = (unsigned short*)(ws);                             // 16 MB
  unsigned short* Kp  = (unsigned short*)(ws + (size_t)16 * 1024 * 1024);  // 4 MB
  unsigned short* Vp  = (unsigned short*)(ws + (size_t)20 * 1024 * 1024);  // 4 MB
  unsigned short* Ctx = (unsigned short*)(ws + (size_t)24 * 1024 * 1024);  // 16 MB

  dim3 blk(256);
  // Projections (fp32 in -> bf16 out)
  gemm_wmma_kernel<false, true><<<dim3(E / 128, M / 128), blk, 0, stream>>>(x, Wq, Qp, M, E, E);
  gemm_wmma_kernel<false, true><<<dim3(KV / 128, M / 128), blk, 0, stream>>>(x, Wk, Kp, M, KV, E);
  gemm_wmma_kernel<false, true><<<dim3(KV / 128, M / 128), blk, 0, stream>>>(x, Wv, Vp, M, KV, E);
  // Flash attention: B*H*(S/128) = 2*32*16 = 1024 workgroups
  attn_wmma_kernel<<<dim3(1024), blk, 0, stream>>>(Qp, Kp, Vp, Ctx);
  // Output projection (bf16 ctx -> fp32 out)
  gemm_wmma_kernel<true, false><<<dim3(E / 128, M / 128), blk, 0, stream>>>(Ctx, Wo, out, M, E, E);
}
